// GraphCastDecoder_77532749627489
// MI455X (gfx1250) — matrix operand order repaired and checked by reading, not compiled, not run on hardware
//
#include <hip/hip_runtime.h>
#include <hip/hip_bf16.h>
#include <stdint.h>

typedef __attribute__((ext_vector_type(16))) __bf16 v16bf;
typedef __attribute__((ext_vector_type(8)))  float  v8f;

#define LDA 528          // 512 + 16 bf16 pad (stagger LDS banks)
#define TILE_ROWS 64

// Dynamic-LDS layout (byte offsets; all LDS is dynamic -> base offset 0)
#define OFF_A    0                         // 64 x 528 bf16 = 67584 B
#define OFF_B1   67584                     // 512 f32
#define OFF_G    69632                     // 512 f32
#define OFF_BE   71680                     // 512 f32
#define OFF_B2   73728                     // 128 f32
#define OFF_SUM  74240                     // 64 f32
#define OFF_SSQ  74496                     // 64 f32
#define OFF_MU   74752                     // 64 f32
#define OFF_RS   75008                     // 64 f32
#define SMEM_BYTES 75264

struct __attribute__((aligned(16))) U128 { unsigned int x, y, z, w; };

union Frag {
  v16bf v;
  U128  q[2];
};

__device__ __forceinline__ unsigned short f2bf(float f) {
  unsigned int u = __float_as_uint(f);
  u += 0x7FFFu + ((u >> 16) & 1u);          // round-to-nearest-even
  return (unsigned short)(u >> 16);
}

// One-time: convert W1 (DxD) and W2 (DxOUT) fp32 row-major -> bf16 column-major
// (n-major, k-contiguous) so B fragments are two contiguous b128 loads per lane.
__global__ void prep_weights_kernel(const float* __restrict__ W1,
                                    const float* __restrict__ W2,
                                    unsigned short* __restrict__ W1T,
                                    unsigned short* __restrict__ W2T,
                                    int D, int OUT) {
  int i = blockIdx.x * blockDim.x + threadIdx.x;
  if (i < D * D) {
    int n = i / D, k = i - n * D;
    W1T[i] = f2bf(W1[k * D + n]);
  }
  if (i < OUT * D) {
    int n = i / D, k = i - n * D;
    W2T[i] = f2bf(W2[k * OUT + n]);
  }
}

__global__ void __launch_bounds__(256)
decoder_fused_kernel(const float* __restrict__ mesh,          // (B, M, 512)
                     const long long* __restrict__ idx,       // (G, 4) int64
                     const float* __restrict__ wts,           // (G, 4)
                     const unsigned short* __restrict__ W1T,  // (512 cols)x(512 k) bf16
                     const unsigned short* __restrict__ W2T,  // (128 cols)x(512 k) bf16
                     const float* __restrict__ b1,
                     const float* __restrict__ gamma,
                     const float* __restrict__ beta,
                     const float* __restrict__ b2,
                     float* __restrict__ out,                 // (R, 128)
                     int R, int G, int Mmesh) {
  extern __shared__ char smem[];
  unsigned short* sh_A  = (unsigned short*)(smem + OFF_A);
  float* sh_b1  = (float*)(smem + OFF_B1);
  float* sh_g   = (float*)(smem + OFF_G);
  float* sh_be  = (float*)(smem + OFF_BE);
  float* sh_b2  = (float*)(smem + OFF_B2);
  float* sh_sum = (float*)(smem + OFF_SUM);
  float* sh_ssq = (float*)(smem + OFF_SSQ);
  float* sh_mu  = (float*)(smem + OFF_MU);
  float* sh_rs  = (float*)(smem + OFF_RS);

  const int tid    = threadIdx.x;
  const int lane   = tid & 31;
  const int wv     = tid >> 5;        // 0..7
  const int lanelo = lane & 15;
  const int hi     = lane >> 4;       // 0 or 1 (half-wave)
  const int rowBase = blockIdx.x * TILE_ROWS;

  // ---- stage per-column params via async copy unit (ASYNCcnt-tracked) ----
  for (int c = tid; c < 512; c += 256) {
    unsigned o1 = OFF_B1 + c * 4; const float* p1 = b1 + c;
    unsigned o2 = OFF_G  + c * 4; const float* p2 = gamma + c;
    unsigned o3 = OFF_BE + c * 4; const float* p3 = beta + c;
    asm volatile("global_load_async_to_lds_b32 %0, %1, off" :: "v"(o1), "v"(p1) : "memory");
    asm volatile("global_load_async_to_lds_b32 %0, %1, off" :: "v"(o2), "v"(p2) : "memory");
    asm volatile("global_load_async_to_lds_b32 %0, %1, off" :: "v"(o3), "v"(p3) : "memory");
  }
  if (tid < 128) {
    unsigned o4 = OFF_B2 + tid * 4; const float* p4 = b2 + tid;
    asm volatile("global_load_async_to_lds_b32 %0, %1, off" :: "v"(o4), "v"(p4) : "memory");
  }
  if (tid < TILE_ROWS) { sh_sum[tid] = 0.f; sh_ssq[tid] = 0.f; }

  // ---- gather + K=4 weighted sum -> sh_A (bf16) ----
  {
    const int r    = tid >> 2;         // 0..63 : row within tile (4 threads/row)
    const int cseg = (tid & 3) * 128;  // 128 columns per thread
    const int row  = rowBase + r;
    unsigned short* dst = sh_A + r * LDA + cseg;
    if (row < R) {
      const int b = row / G;
      const int g = row - b * G;
      const long long* ip = idx + (long long)g * 4;
      const float*     wp = wts + (long long)g * 4;
      const float w0 = wp[0], w1 = wp[1], w2 = wp[2], w3 = wp[3];
      const float* p0 = mesh + ((long long)b * Mmesh + ip[0]) * 512 + cseg;
      const float* p1 = mesh + ((long long)b * Mmesh + ip[1]) * 512 + cseg;
      const float* p2 = mesh + ((long long)b * Mmesh + ip[2]) * 512 + cseg;
      const float* p3 = mesh + ((long long)b * Mmesh + ip[3]) * 512 + cseg;
      #pragma unroll 4
      for (int c4 = 0; c4 < 32; ++c4) {
        float4 a0 = ((const float4*)p0)[c4];
        float4 a1 = ((const float4*)p1)[c4];
        float4 a2 = ((const float4*)p2)[c4];
        float4 a3 = ((const float4*)p3)[c4];
        float sx = w0 * a0.x + w1 * a1.x + w2 * a2.x + w3 * a3.x;
        float sy = w0 * a0.y + w1 * a1.y + w2 * a2.y + w3 * a3.y;
        float sz = w0 * a0.z + w1 * a1.z + w2 * a2.z + w3 * a3.z;
        float sw = w0 * a0.w + w1 * a1.w + w2 * a2.w + w3 * a3.w;
        unsigned long long pk =
            (unsigned long long)((unsigned int)f2bf(sx) | ((unsigned int)f2bf(sy) << 16)) |
            ((unsigned long long)((unsigned int)f2bf(sz) | ((unsigned int)f2bf(sw) << 16)) << 32);
        *(unsigned long long*)(dst + c4 * 4) = pk;
      }
    } else {
      #pragma unroll 4
      for (int c4 = 0; c4 < 32; ++c4)
        *(unsigned long long*)(dst + c4 * 4) = 0ull;
    }
  }
  asm volatile("s_wait_asynccnt 0x0" ::: "memory");
  __syncthreads();

  // warm W2T into caches while GEMM1 runs
  __builtin_prefetch(W2T + (size_t)(wv * 16 + lanelo) * 512, 0, 1);

  // ---- GEMM1: (64x512) @ W1 (512x512); wave owns 64 cols, 4 M-subtiles ----
  const int colW = wv * 64;
  v8f zero = {0.f, 0.f, 0.f, 0.f, 0.f, 0.f, 0.f, 0.f};
  v8f acc[4][4];
  #pragma unroll
  for (int i = 0; i < 4; ++i)
    #pragma unroll
    for (int j = 0; j < 4; ++j) acc[i][j] = zero;

  for (int ks = 0; ks < 16; ++ks) {
    const int k0 = ks * 32;
    Frag a[4];
    #pragma unroll
    for (int i = 0; i < 4; ++i) {
      // A layout: lanes 0-15: K [k0,k0+8) + [k0+16,k0+24); lanes 16-31: +8
      const unsigned short* src = sh_A + (i * 16 + lanelo) * LDA + k0 + hi * 8;
      a[i].q[0] = *(const U128*)(src);
      a[i].q[1] = *(const U128*)(src + 16);
    }
    #pragma unroll
    for (int j = 0; j < 4; ++j) {
      // B layout: lane n holds col n, K [k0,k0+8) + [k0+16,k0+24) (lo half) / +8 (hi half)
      const unsigned short* bsrc = W1T + (size_t)(colW + j * 16 + lanelo) * 512 + k0 + hi * 8;
      Frag bf;
      bf.q[0] = ((const U128*)bsrc)[0];
      bf.q[1] = ((const U128*)bsrc)[2];
      #pragma unroll
      for (int i = 0; i < 4; ++i)
        acc[i][j] = __builtin_amdgcn_wmma_f32_16x16x32_bf16(
            false, a[i].v, false, bf.v, (short)0, acc[i][j], false, false);
    }
  }

  // ---- +b1, LayerNorm stats (cross-wave via LDS float atomics) ----
  #pragma unroll
  for (int i = 0; i < 4; ++i) {
    #pragma unroll
    for (int v = 0; v < 8; ++v) {
      float s = 0.f, s2 = 0.f;
      #pragma unroll
      for (int j = 0; j < 4; ++j) {
        const int col = colW + j * 16 + lanelo;
        float val = acc[i][j][v] + sh_b1[col];
        acc[i][j][v] = val;
        s += val; s2 += val * val;
      }
      const int row = i * 16 + v + hi * 8;   // C/D layout: VGPR v -> M=v (+8 for hi half)
      atomicAdd(&sh_sum[row], s);
      atomicAdd(&sh_ssq[row], s2);
    }
  }
  __syncthreads();

  if (tid < TILE_ROWS) {
    const float inv = 1.0f / 512.0f;
    float mu  = sh_sum[tid] * inv;
    float var = sh_ssq[tid] * inv - mu * mu;
    sh_mu[tid] = mu;
    sh_rs[tid] = rsqrtf(var + 1e-5f);
  }
  __syncthreads();

  // ---- normalize + affine + SiLU -> reuse sh_A as h (bf16) ----
  #pragma unroll
  for (int i = 0; i < 4; ++i) {
    #pragma unroll
    for (int v = 0; v < 8; ++v) {
      const int row = i * 16 + v + hi * 8;
      const float mu = sh_mu[row], rs = sh_rs[row];
      #pragma unroll
      for (int j = 0; j < 4; ++j) {
        const int col = colW + j * 16 + lanelo;
        float hv = (acc[i][j][v] - mu) * rs * sh_g[col] + sh_be[col];
        hv = hv / (1.0f + __expf(-hv));       // SiLU
        sh_A[row * LDA + col] = f2bf(hv);
      }
    }
  }
  __syncthreads();

  // ---- GEMM2: h (64x512) @ W2 (512x128); wave owns 16 output cols ----
  v8f acc2[4] = { zero, zero, zero, zero };
  for (int ks = 0; ks < 16; ++ks) {
    const int k0 = ks * 32;
    Frag a[4], bf;
    #pragma unroll
    for (int i = 0; i < 4; ++i) {
      const unsigned short* src = sh_A + (i * 16 + lanelo) * LDA + k0 + hi * 8;
      a[i].q[0] = *(const U128*)(src);
      a[i].q[1] = *(const U128*)(src + 16);
    }
    const unsigned short* bsrc = W2T + (size_t)(wv * 16 + lanelo) * 512 + k0 + hi * 8;
    bf.q[0] = ((const U128*)bsrc)[0];
    bf.q[1] = ((const U128*)bsrc)[2];
    #pragma unroll
    for (int i = 0; i < 4; ++i)
      acc2[i] = __builtin_amdgcn_wmma_f32_16x16x32_bf16(
          false, a[i].v, false, bf.v, (short)0, acc2[i], false, false);
  }

  // ---- epilogue: + b2, store fp32 ----
  #pragma unroll
  for (int i = 0; i < 4; ++i) {
    #pragma unroll
    for (int v = 0; v < 8; ++v) {
      const int row = rowBase + i * 16 + v + hi * 8;
      if (row < R) {
        const int col = wv * 16 + lanelo;
        out[(long long)row * 128 + col] = acc2[i][v] + sh_b2[col];
      }
    }
  }
}

extern "C" void kernel_launch(void* const* d_in, const int* in_sizes, int n_in,
                              void* d_out, int out_size, void* d_ws, size_t ws_size,
                              hipStream_t stream) {
  const float*     mesh  = (const float*)d_in[0];
  const long long* idx   = (const long long*)d_in[1];
  const float*     wts   = (const float*)d_in[2];
  const float*     W1    = (const float*)d_in[3];
  const float*     b1    = (const float*)d_in[4];
  const float*     gamma = (const float*)d_in[5];
  const float*     beta  = (const float*)d_in[6];
  const float*     W2    = (const float*)d_in[7];
  const float*     b2    = (const float*)d_in[8];
  float*           out   = (float*)d_out;

  const int D = 512, OUT = 128, Kg = 4;
  const int G = in_sizes[1] / Kg;               // 65160
  const int B = out_size / (G * OUT);           // 2
  const int Mmesh = in_sizes[0] / (B * D);      // 40962
  const int R = B * G;                          // 130320

  unsigned short* W1T = (unsigned short*)d_ws;
  unsigned short* W2T = W1T + (size_t)D * D;

  const int prepN = D * D;
  prep_weights_kernel<<<(prepN + 255) / 256, 256, 0, stream>>>(W1, W2, W1T, W2T, D, OUT);

  const int nblocks = (R + TILE_ROWS - 1) / TILE_ROWS;
  decoder_fused_kernel<<<nblocks, 256, SMEM_BYTES, stream>>>(
      mesh, idx, wts, W1T, W2T, b1, gamma, beta, b2, out, R, G, Mmesh);
}
